// GraphContext_240518168853
// MI455X (gfx1250) — matrix operand order repaired and checked by reading, not compiled, not run on hardware
//
#include <hip/hip_runtime.h>

#define C 1024
#define Kp 16
#define D 512
#define HID 1024
#define B 65536
#define PSTR 516   // padded LDS row stride (floats): 16B-aligned chunks, conflict-free

typedef float  v8f   __attribute__((ext_vector_type(8)));
typedef float  v4f   __attribute__((ext_vector_type(4)));
typedef int    v4i   __attribute__((ext_vector_type(4)));
typedef __bf16 v16bf __attribute__((ext_vector_type(16)));
typedef __bf16 v8bf  __attribute__((ext_vector_type(8)));

#if defined(__has_builtin)
#if __has_builtin(__builtin_amdgcn_global_load_async_to_lds_b128) && \
    __has_builtin(__builtin_amdgcn_s_wait_asynccnt)
#define HAVE_ASYNC_LDS 1
#endif
#endif

// 16-byte global -> LDS copy (async on gfx1250 when available)
__device__ __forceinline__ void cp16(void* dst_lds, const void* src_glb) {
#ifdef HAVE_ASYNC_LDS
    __builtin_amdgcn_global_load_async_to_lds_b128(
        (__attribute__((address_space(1))) v4i*)(size_t)src_glb,
        (__attribute__((address_space(3))) v4i*)dst_lds, 0, 0);
#else
    *(v4f*)dst_lds = *(const v4f*)src_glb;
#endif
}
__device__ __forceinline__ void cp16_wait() {
#ifdef HAVE_ASYNC_LDS
    __builtin_amdgcn_s_wait_asynccnt(0);
#endif
}

// ---------------- weight conversion: f32 -> bf16, transposed [N][K] ----------
__global__ void k_convert_w(const float* __restrict__ w1, const float* __restrict__ w2,
                            __bf16* __restrict__ w1t, __bf16* __restrict__ w2t) {
    int idx = blockIdx.x * 256 + threadIdx.x;          // 0 .. 2*524288-1
    if (idx < D * HID) {
        int d = idx / HID, h = idx % HID;              // w1 is [D,HID]
        w1t[h * D + d] = (__bf16)w1[idx];              // w1t is [HID,D]
    } else {
        int i2 = idx - D * HID;
        int h = i2 / D, d = i2 % D;                    // w2 is [HID,D]
        w2t[d * HID + h] = (__bf16)w2[i2];             // w2t is [D,HID]
    }
}

// ---------------- per-cell graph: Gram -> Ahat, Lsym, H0 = Ahat @ protos -----
__global__ void k_cell_graph(const float* __restrict__ protos,
                             float* __restrict__ Ahat, float* __restrict__ Lsym,
                             __bf16* __restrict__ H0bf) {
    const int c = blockIdx.x, t = threadIdx.x;
    __shared__ float P[16 * PSTR];
    __shared__ float Wm[256], Ah[256];
    __shared__ float x2s[16], diss[16], dAs[16], disAs[16];
    const float* pc = protos + (size_t)c * (Kp * D);
    for (int i4 = t * 4; i4 < Kp * D; i4 += 1024) {    // async 16B chunks
        int k = i4 >> 9, d = i4 & 511;
        cp16(&P[k * PSTR + d], pc + i4);
    }
    cp16_wait();
    __syncthreads();
    const int k = t >> 4, j = t & 15;
    const float* pk = &P[k * PSTR];
    const float* pj = &P[j * PSTR];
    float dot = 0.f;
    for (int d = 0; d < D; ++d) dot += pk[d] * pj[d];
    if (j == k) x2s[k] = dot;
    __syncthreads();
    float d2 = fmaxf(x2s[k] + x2s[j] - 2.f * dot, 0.f);
    float w  = __expf(-d2) + (k == j ? 1.f : 0.f);     // TAU = 1
    Wm[t] = w;
    __syncthreads();
    if (t < 16) { float s = 0.f; for (int jj = 0; jj < 16; ++jj) s += Wm[t * 16 + jj];
                  diss[t] = rsqrtf(fmaxf(s, 1e-6f)); }
    __syncthreads();
    float a = diss[k] * w * diss[j];
    Ah[t] = a; Ahat[(size_t)c * 256 + t] = a;
    __syncthreads();
    if (t < 16) { float s = 0.f; for (int jj = 0; jj < 16; ++jj) s += Ah[t * 16 + jj];
                  dAs[t] = s; disAs[t] = rsqrtf(fmaxf(s, 1e-6f)); }
    __syncthreads();
    float l = (k == j ? dAs[k] : 0.f) - a;
    Lsym[(size_t)c * 256 + t] = disAs[k] * l * disAs[j];
    // H0 = Ahat @ P  (16 x 512), packed bf16 stores
    for (int base = t * 8; base < Kp * D; base += 2048) {
        int kk = base >> 9, d = base & 511;
        v8bf o;
#pragma unroll
        for (int e = 0; e < 8; ++e) {
            float s = 0.f;
            for (int jj = 0; jj < 16; ++jj) s += Ah[kk * 16 + jj] * P[jj * PSTR + d + e];
            o[e] = (__bf16)s;
        }
        *(v8bf*)(H0bf + (size_t)c * (Kp * D) + base) = o;
    }
}

// ---------------- bf16 WMMA GEMM: out = A[M,K] @ Bt[N,K]^T -------------------
// one wave computes a 16x64 strip; ping-pong double-buffered fragments
__global__ void k_gemm(const __bf16* __restrict__ A, const __bf16* __restrict__ Bt,
                       float* __restrict__ outF, __bf16* __restrict__ outBf,
                       int N, int K, int doRelu) {
    const int wave = (blockIdx.x * blockDim.x + threadIdx.x) >> 5;
    const int lane = threadIdx.x & 31;
    const int groupsN = N >> 6;
    const int tm = wave / groupsN, tg = wave % groupsN;
    const int m0 = tm << 4, n0 = tg << 6;
    const int h = lane >> 4, ln = lane & 15;
    const __bf16* arow = A + (size_t)(m0 + ln) * K;
    const __bf16* bbase[4];
#pragma unroll
    for (int tn = 0; tn < 4; ++tn)
        bbase[tn] = Bt + (size_t)(n0 + tn * 16 + ln) * K + 16 * h;

    union AB { v16bf v; v8bf p[2]; };
    AB a0, a1, b0[4], b1[4];
    v8f acc[4];
#pragma unroll
    for (int i = 0; i < 4; ++i)
#pragma unroll
        for (int e = 0; e < 8; ++e) acc[i][e] = 0.f;

#define LOADA(k0, dst) { (dst).p[0] = *(const v8bf*)(arow + (k0) + 16 * h); \
                         (dst).p[1] = *(const v8bf*)(arow + (k0) + 16 * h + 8); }
#define LOADB(k0, tn, dst) { const __bf16* br = bbase[tn] + (k0); \
                             (dst).p[0] = *(const v8bf*)br; \
                             (dst).p[1] = *(const v8bf*)(br + 8); }

    LOADA(0, a0);
#pragma unroll
    for (int tn = 0; tn < 4; ++tn) LOADB(0, tn, b0[tn]);

    for (int k0 = 0; k0 < K; k0 += 64) {               // K is a multiple of 64
        LOADA(k0 + 32, a1);
#pragma unroll
        for (int tn = 0; tn < 4; ++tn) LOADB(k0 + 32, tn, b1[tn]);
#pragma unroll
        for (int tn = 0; tn < 4; ++tn)
            acc[tn] = __builtin_amdgcn_wmma_f32_16x16x32_bf16(
                false, a0.v, false, b0[tn].v, (short)0, acc[tn], false, false);
        if (k0 + 64 < K) {
            LOADA(k0 + 64, a0);
#pragma unroll
            for (int tn = 0; tn < 4; ++tn) LOADB(k0 + 64, tn, b0[tn]);
        }
#pragma unroll
        for (int tn = 0; tn < 4; ++tn)
            acc[tn] = __builtin_amdgcn_wmma_f32_16x16x32_bf16(
                false, a1.v, false, b1[tn].v, (short)0, acc[tn], false, false);
    }
#undef LOADA
#undef LOADB

#pragma unroll
    for (int tn = 0; tn < 4; ++tn) {
        int col = n0 + tn * 16 + ln;
#pragma unroll
        for (int r = 0; r < 8; ++r) {
            int row = m0 + r + 8 * h;                  // C/D layout: VGPR r -> M=r+8h
            float v = acc[tn][r];
            if (doRelu) v = fmaxf(v, 0.f);
            if (outF)  outF [(size_t)row * N + col] = v;
            if (outBf) outBf[(size_t)row * N + col] = (__bf16)v;
        }
    }
}

// ---------------- per-cell propagation: H2 = Ahat @ H1 -----------------------
__global__ void k_prop(const float* __restrict__ Ahat, const __bf16* __restrict__ H1,
                       __bf16* __restrict__ H2) {
    const int c = blockIdx.x, t = threadIdx.x;
    __shared__ float Ah[256];
    Ah[t] = Ahat[(size_t)c * 256 + t];
    __syncthreads();
    const __bf16* h1 = H1 + (size_t)c * (Kp * HID);
    __bf16* h2 = H2 + (size_t)c * (Kp * HID);
    for (int base = t * 8; base < Kp * HID; base += 2048) {
        int k = base >> 10, hh = base & 1023;
        float s[8] = {0.f, 0.f, 0.f, 0.f, 0.f, 0.f, 0.f, 0.f};
        for (int jj = 0; jj < 16; ++jj) {
            float w = Ah[k * 16 + jj];
            v8bf hv = *(const v8bf*)(h1 + jj * HID + hh);
#pragma unroll
            for (int e = 0; e < 8; ++e) s[e] += w * (float)hv[e];
        }
        v8bf o;
#pragma unroll
        for (int e = 0; e < 8; ++e) o[e] = (__bf16)s[e];
        *(v8bf*)(h2 + base) = o;
    }
}

// ---------------- mu = mean_k H ----------------------------------------------
__global__ void k_mu(const float* __restrict__ H, float* __restrict__ mu) {
    int idx = blockIdx.x * 256 + threadIdx.x;          // < C*D
    int c = idx >> 9, d = idx & 511;
    const float* h = H + (size_t)c * (Kp * D) + d;
    float s = 0.f;
    for (int k = 0; k < Kp; ++k) s += h[k * D];
    mu[idx] = s * (1.f / 16.f);
}

// ---------------- smooth partials: sum_{k,j,d} H Lsym H per cell -------------
__global__ void k_smooth(const float* __restrict__ H, const float* __restrict__ Lsym,
                         float* __restrict__ part) {
    const int c = blockIdx.x, t = threadIdx.x;
    __shared__ float Hs[16 * PSTR];
    __shared__ float Ls[256];
    __shared__ float red[256];
    Ls[t] = Lsym[(size_t)c * 256 + t];
    const float* hc = H + (size_t)c * (Kp * D);
    for (int i4 = t * 4; i4 < Kp * D; i4 += 1024) {    // async 16B chunks
        int k = i4 >> 9, d = i4 & 511;
        cp16(&Hs[k * PSTR + d], hc + i4);
    }
    cp16_wait();
    __syncthreads();
    float acc = 0.f;
    for (int i = t; i < Kp * D; i += 256) {
        int k = i >> 9, d = i & 511;
        float s = 0.f;
        for (int jj = 0; jj < 16; ++jj) s += Ls[k * 16 + jj] * Hs[jj * PSTR + d];
        acc += s * Hs[k * PSTR + d];
    }
    red[t] = acc; __syncthreads();
    for (int s = 128; s > 0; s >>= 1) { if (t < s) red[t] += red[t + s]; __syncthreads(); }
    if (t == 0) part[c] = red[0];
}

// ---------------- margin partials: hinge on pairwise mu distances ------------
__global__ void k_margin(const float* __restrict__ mu, float* __restrict__ part) {
    const int i = blockIdx.x, t = threadIdx.x;
    __shared__ float mi[512];
    __shared__ float red[256];
    if (t < 128) cp16(&mi[t * 4], mu + (size_t)i * D + t * 4);
    cp16_wait();
    __syncthreads();
    float m2i = 0.f;
    for (int d = 0; d < D; ++d) m2i += mi[d] * mi[d];
    float acc = 0.f;
    for (int jj = t; jj < C; jj += 256) {
        if (jj == i) continue;
        const float* mj = mu + (size_t)jj * D;
        float dotv = 0.f, m2j = 0.f;
        for (int d = 0; d < D; ++d) { float v = mj[d]; dotv += mi[d] * v; m2j += v * v; }
        float dd = fmaxf(m2i + m2j - 2.f * dotv, 1e-12f);
        acc += fmaxf(1.0f - sqrtf(dd), 0.f);           // MARGIN = 1
    }
    red[t] = acc; __syncthreads();
    for (int s = 128; s > 0; s >>= 1) { if (t < s) red[t] += red[t + s]; __syncthreads(); }
    if (t == 0) part[i] = red[0];
}

// ---------------- gather mu_b into d_out + center partials -------------------
__global__ void k_center(const float* __restrict__ zF, const int* __restrict__ labels,
                         const float* __restrict__ mu, float* __restrict__ out,
                         float* __restrict__ part) {
    int idx = blockIdx.x * 256 + threadIdx.x;          // < B*D
    int b = idx >> 9, d = idx & 511;
    int lab = labels[b];
    float m = mu[(size_t)lab * D + d];
    out[idx] = m;
    float diff = zF[idx] - m;
    __shared__ float red[256];
    red[threadIdx.x] = diff * diff; __syncthreads();
    for (int s = 128; s > 0; s >>= 1) { if (threadIdx.x < s) red[threadIdx.x] += red[threadIdx.x + s]; __syncthreads(); }
    if (threadIdx.x == 0) part[blockIdx.x] = red[0];
}

// ---------------- deterministic scalar reduction ------------------------------
__global__ void k_reduce_scale(const float* __restrict__ in, int n, float scale,
                               float* __restrict__ out) {
    __shared__ float red[256];
    int t = threadIdx.x;
    float s = 0.f;
    for (int i = t; i < n; i += 256) s += in[i];
    red[t] = s; __syncthreads();
    for (int w = 128; w > 0; w >>= 1) { if (t < w) red[t] += red[t + w]; __syncthreads(); }
    if (t == 0) *out = red[0] * scale;
}

extern "C" void kernel_launch(void* const* d_in, const int* in_sizes, int n_in,
                              void* d_out, int out_size, void* d_ws, size_t ws_size,
                              hipStream_t stream) {
    (void)in_sizes; (void)n_in; (void)out_size; (void)ws_size;
    const float* zF     = (const float*)d_in[0];
    const int*   labels = (const int*)  d_in[1];
    const float* protos = (const float*)d_in[2];
    const float* w1     = (const float*)d_in[3];
    const float* w2     = (const float*)d_in[4];
    float* out = (float*)d_out;

    char* p = (char*)d_ws;
    auto alloc = [&](size_t bytes) -> char* {
        char* r = p; p += (bytes + 255) & ~(size_t)255; return r;
    };
    float*  Ahat       = (float*) alloc((size_t)C * 256 * 4);
    float*  Lsym       = (float*) alloc((size_t)C * 256 * 4);
    float*  mu         = (float*) alloc((size_t)C * D * 4);
    float*  Hf         = (float*) alloc((size_t)C * Kp * D * 4);
    __bf16* H0bf       = (__bf16*)alloc((size_t)C * Kp * D * 2);
    __bf16* w1t        = (__bf16*)alloc((size_t)HID * D * 2);
    __bf16* w2t        = (__bf16*)alloc((size_t)D * HID * 2);
    __bf16* H1bf       = (__bf16*)alloc((size_t)C * Kp * HID * 2);
    __bf16* H2bf       = (__bf16*)alloc((size_t)C * Kp * HID * 2);
    float*  smoothPart = (float*) alloc((size_t)C * 4);
    float*  marginPart = (float*) alloc((size_t)C * 4);
    const int centerBlocks = (B * D) / 256;            // 131072
    float*  centerPart = (float*) alloc((size_t)centerBlocks * 4);

    // 1) weights -> bf16 (transposed [N][K])
    k_convert_w<<<(2 * D * HID) / 256, 256, 0, stream>>>(w1, w2, w1t, w2t);
    // 2) per-cell graph + H0
    k_cell_graph<<<C, 256, 0, stream>>>(protos, Ahat, Lsym, H0bf);
    // 3) H1 = relu(H0 @ w1)   [16384,512]x[512,1024]
    k_gemm<<<(C * Kp / 16) * (HID / 64) / 8, 256, 0, stream>>>(H0bf, w1t, nullptr, H1bf, HID, D, 1);
    // 4) H2 = Ahat @ H1 (per cell)
    k_prop<<<C, 256, 0, stream>>>(Ahat, H1bf, H2bf);
    // 5) H = H2 @ w2          [16384,1024]x[1024,512]
    k_gemm<<<(C * Kp / 16) * (D / 64) / 8, 256, 0, stream>>>(H2bf, w2t, Hf, nullptr, D, HID, 0);
    // 6) mu = mean_k H
    k_mu<<<(C * D) / 256, 256, 0, stream>>>(Hf, mu);
    // 7) smooth partials
    k_smooth<<<C, 256, 0, stream>>>(Hf, Lsym, smoothPart);
    // 8) margin partials
    k_margin<<<C, 256, 0, stream>>>(mu, marginPart);
    // 9) mu_b gather + center partials
    k_center<<<centerBlocks, 256, 0, stream>>>(zF, labels, mu, out, centerPart);
    // 10) scalars: (smooth, center, margin) after mu_b
    float* scalars = out + (size_t)B * D;
    k_reduce_scale<<<1, 256, 0, stream>>>(smoothPart, C, 1.f / (float)(C * Kp), scalars + 0);
    k_reduce_scale<<<1, 256, 0, stream>>>(centerPart, centerBlocks, 1.f / (float)((size_t)B * D), scalars + 1);
    k_reduce_scale<<<1, 256, 0, stream>>>(marginPart, C, 1.f / ((float)C * (float)(C - 1)), scalars + 2);
}
